// MultiDomainLoss_90503550861918
// MI455X (gfx1250) — matrix operand order repaired and checked by reading, not compiled, not run on hardware
//
#include <hip/hip_runtime.h>
#include <hip/hip_bf16.h>

// ---------------------------------------------------------------------------
// MultiDomainLoss for MI455X (gfx1250, wave32).
//  - soft-DTW: one workgroup per batch, wavefront over 2T-1 diagonals, rows in LDS
//  - CDF cumsum implemented as H @ UpperTriOnes via v_wmma_f32_16x16x32_f16
// ---------------------------------------------------------------------------

#define BIGV        10000000.0f
#define INV_GAMMA   10.0f          // 1 / 0.1
#define GAMMA_DTW   0.1f
#define NBINS       64
#define SIGMA_H     0.05f
#define NB          64
#define NT          1024
#define NTSD        1022
#define NLVL        5

typedef __attribute__((ext_vector_type(16))) _Float16 v16h;
typedef __attribute__((ext_vector_type(8)))  float    v8f;

__constant__ float c_db4_lo[8] = {
    -0.010597401784997278f,  0.032883011666982945f,  0.030841381835986965f,
    -0.18703481171888114f,  -0.02798376941698385f,   0.6308807679295904f,
     0.7148465705525415f,    0.23037781330885523f };
__constant__ float c_db4_hi[8] = {
     0.23037781330885523f,  -0.7148465705525415f,    0.6308807679295904f,
     0.02798376941698385f,  -0.18703481171888114f,  -0.030841381835986965f,
     0.032883011666982945f,  0.010597401784997278f };

// ---------------------------------------------------------------- init ------
__global__ void md_init_acc(float* acc) {
    if (threadIdx.x < 16) acc[threadIdx.x] = 0.0f;
}

// ----------------------------------------------------- second derivative ----
__global__ void md_sd_kernel(const float* __restrict__ p, const float* __restrict__ t,
                             float* __restrict__ sdp, float* __restrict__ sdt) {
    int idx = blockIdx.x * 256 + threadIdx.x;
    if (idx < NB * NTSD) {
        int b = idx / NTSD, i = idx - b * NTSD;
        const float* x = p + (size_t)b * NT;
        const float* y = t + (size_t)b * NT;
        sdp[idx] = x[i + 2] - 2.0f * x[i + 1] + x[i];
        sdt[idx] = y[i + 2] - 2.0f * y[i + 1] + y[i];
    }
}

// ------------------------------------------------------------- soft-DTW -----
// One block per batch. n <= 1024. Rows of length n+1 rotate through LDS.
__global__ __launch_bounds__(1024) void md_dtw_kernel(
        const float* __restrict__ X, const float* __restrict__ Y,
        int n, int stride, float* __restrict__ out) {
    __shared__ float xs[NT];
    __shared__ float ys[NT];
    __shared__ float rows[3 * (NT + 1)];

    const int b   = blockIdx.x;
    const int tid = threadIdx.x;
    const int bd  = blockDim.x;

    for (int i = tid; i < n; i += bd) {
        __builtin_prefetch(X + (size_t)b * stride + i, 0, 0);
        xs[i] = X[(size_t)b * stride + i];
        ys[i] = Y[(size_t)b * stride + i];
    }
    float* r0 = rows;
    float* r1 = rows + (n + 1);
    float* r2 = rows + 2 * (n + 1);
    for (int a = tid; a <= n; a += bd) {
        r0[a] = (a == 0) ? 0.0f : BIGV;   // d0
        r1[a] = BIGV;                     // d1
    }
    __syncthreads();

    float* dm2 = r0;
    float* dm1 = r1;
    float* cur = r2;
    const int kmax = 2 * n;
    for (int k = 2; k <= kmax; ++k) {
        for (int a = tid; a <= n; a += bd) {
            int j = k - a;
            float v = BIGV;
            if (a >= 1 && j >= 1 && j <= n) {
                float up   = dm1[a - 1];   // R[i-1][j]
                float left = dm1[a];       // R[i][j-1]
                float diag = dm2[a - 1];   // R[i-1][j-1]
                float m = fminf(up, fminf(left, diag));
                float s = __expf((m - up)   * INV_GAMMA)
                        + __expf((m - left) * INV_GAMMA)
                        + __expf((m - diag) * INV_GAMMA);
                float softmin = m - GAMMA_DTW * __logf(s);
                float d = xs[a - 1] - ys[j - 1];
                v = d * d + softmin;
            }
            cur[a] = v;
        }
        __syncthreads();
        float* tmp = dm2; dm2 = dm1; dm1 = cur; cur = tmp;
    }
    if (tid == 0) out[b] = dm1[n];
}

// ------------------------------------------------------------ abs mean ------
__global__ void md_absmean_kernel(const float* __restrict__ src, int n,
                                  float* __restrict__ acc, float inv_n) {
    __shared__ float red[256];
    int tid = threadIdx.x;
    float s = 0.0f;
    for (int i = blockIdx.x * 256 + tid; i < n; i += gridDim.x * 256)
        s += fabsf(src[i]);
    red[tid] = s; __syncthreads();
    for (int st = 128; st > 0; st >>= 1) {
        if (tid < st) red[tid] += red[tid + st];
        __syncthreads();
    }
    if (tid == 0) atomicAdd(acc, red[0] * inv_n);
}

// ------------------------------------------------------ db4 wavelet map -----
// One block per batch; 5 levels of stride-2 cross-correlation, SAME pad (3,3).
__global__ void md_wavelet_kernel(const float* __restrict__ x, float* __restrict__ fmap) {
    __shared__ float bufA[NT];
    __shared__ float bufB[NT];
    const int b = blockIdx.x, tid = threadIdx.x;
    for (int i = tid; i < NT; i += 256) bufA[i] = x[(size_t)b * NT + i];
    __syncthreads();
    float* curb = bufA;
    float* nxtb = bufB;
    int N = NT;
    for (int lvl = 0; lvl < NLVL; ++lvl) {
        int M = N >> 1;
        float* out = fmap + ((size_t)b * NLVL + lvl) * NT;
        for (int o = tid; o < NT; o += 256) {
            if (o < M) {
                float dd = 0.0f, aa = 0.0f;
#pragma unroll
                for (int k = 0; k < 8; ++k) {
                    int ii = 2 * o + k - 3;
                    float v = (ii >= 0 && ii < N) ? curb[ii] : 0.0f;
                    dd += v * c_db4_hi[k];
                    aa += v * c_db4_lo[k];
                }
                nxtb[o] = aa;
                out[o]  = dd;
            } else {
                out[o] = 0.0f;   // pad to T with zeros
            }
        }
        __syncthreads();
        float* t2 = curb; curb = nxtb; nxtb = t2;
        N = M;
    }
}

// ------------------------------------------------- frequency sparsity -------
__global__ void md_freq_kernel(const float* __restrict__ fmap_p, float* __restrict__ acc) {
    __shared__ float red[256];
    __shared__ float energy[NLVL];
    const int b = blockIdx.x, tid = threadIdx.x;
    for (int lvl = 0; lvl < NLVL; ++lvl) {
        const float* row = fmap_p + ((size_t)b * NLVL + lvl) * NT;
        float s = 0.0f;
        for (int i = tid; i < NT; i += 256) s += fabsf(row[i]);
        red[tid] = s; __syncthreads();
        for (int st = 128; st > 0; st >>= 1) {
            if (tid < st) red[tid] += red[tid + st];
            __syncthreads();
        }
        if (tid == 0) energy[lvl] = red[0] * (1.0f / (float)NT);
        __syncthreads();
    }
    if (tid == 0) {
        int peak = 0; float best = energy[0];
        for (int l = 1; l < NLVL; ++l)
            if (energy[l] > best) { best = energy[l]; peak = l; }
        float num = 0.0f, den = 0.0f;
        for (int l = 0; l < NLVL; ++l) {
            den += energy[l];
            if (l >= peak - 2 && l <= peak + 2) num += energy[l];
        }
        float sel = num / (den + 1e-6f);
        atomicAdd(acc, (1.0f - sel) * (1.0f / (float)NB));
    }
}

// ----------------------------------------------------- soft histogram -------
// One block per batch-row. Normalized (per-row min/max) soft histogram of 64
// Gaussian bins; writes normalized h (sums to 1) to hout[b*64 ..].
__global__ void md_hist_kernel(const float* __restrict__ src, int N, int stride,
                               float* __restrict__ hout) {
    __shared__ float zs[5 * NT];      // up to 5120
    __shared__ float red[256];
    __shared__ float hsh[NBINS];
    __shared__ float lohi[2];
    const int b = blockIdx.x, tid = threadIdx.x;
    const float* row = src + (size_t)b * stride;
    float lo = 3.4e38f, hv = -3.4e38f;
    for (int i = tid; i < N; i += 256) {
        float v = row[i];
        zs[i] = v;
        lo = fminf(lo, v);
        hv = fmaxf(hv, v);
    }
    red[tid] = lo; __syncthreads();
    for (int st = 128; st > 0; st >>= 1) { if (tid < st) red[tid] = fminf(red[tid], red[tid + st]); __syncthreads(); }
    if (tid == 0) lohi[0] = red[0];
    __syncthreads();
    red[tid] = hv; __syncthreads();
    for (int st = 128; st > 0; st >>= 1) { if (tid < st) red[tid] = fmaxf(red[tid], red[tid + st]); __syncthreads(); }
    if (tid == 0) lohi[1] = red[0];
    __syncthreads();

    const float lo_   = lohi[0];
    const float scale = 1.0f / (lohi[1] - lo_ + 1e-6f);
    const int   c     = tid & 63;
    const int   sub   = tid >> 6;                    // 4 partials per center
    const float center = (float)c * (1.0f / 63.0f);  // linspace(0,1,64)
    const float invs  = 1.0f / SIGMA_H;
    float accp = 0.0f;
    for (int i = sub; i < N; i += 4) {
        float z = (zs[i] - lo_) * scale;
        float d = (z - center) * invs;
        accp += __expf(-0.5f * d * d);
    }
    red[tid] = accp; __syncthreads();
    if (tid < NBINS)
        hsh[tid] = red[tid] + red[tid + 64] + red[tid + 128] + red[tid + 192] + 1e-6f;
    __syncthreads();
    if (tid < NBINS) red[tid] = hsh[tid]; else red[tid] = 0.0f;
    __syncthreads();
    for (int st = 32; st > 0; st >>= 1) { if (tid < st) red[tid] += red[tid + st]; __syncthreads(); }
    const float invtot = 1.0f / red[0];
    if (tid < NBINS) hout[(size_t)b * NBINS + tid] = hsh[tid] * invtot;
}

// ------------------------------------------------ CDF cumsum via WMMA -------
// CDF = H(nrows x 64) @ U(64 x 64), U[i][j] = (i <= j). One wave32 per 16x16
// output tile; K = 64 -> two v_wmma_f32_16x16x32_f16 issues per tile.
__global__ __launch_bounds__(32) void md_cumsum_wmma(const float* __restrict__ h,
                                                     float* __restrict__ cdf) {
    const int tile  = blockIdx.x;
    const int mtile = tile >> 2;        // nrows/16 tiles
    const int ntile = tile & 3;         // 64/16 = 4 tiles
    const int lane  = threadIdx.x;
    const int l15   = lane & 15;
    const int hi    = lane >> 4;

    v8f c = {};
#pragma unroll
    for (int kstep = 0; kstep < 2; ++kstep) {
        v16h a, bf;
        const int row = mtile * 16 + l15;
        const int n   = ntile * 16 + l15;
#pragma unroll
        for (int t = 0; t < 16; ++t) {
            // A layout (16-bit, 16x32): lanes 0-15 hold K 0-7 (V0-3) & 16-23 (V4-7),
            // lanes 16-31 hold K 8-15 & 24-31.
            int ka = kstep * 32 + t + hi * 8 + ((t >= 8) ? 8 : 0);
            a[t] = (_Float16)h[(size_t)row * NBINS + ka];
            // B layout (16-bit, 32x16): lanes 0-15 hold K 0-7 & 16-23, lanes 16-31 K 8-15 & 24-31.
            int kb = kstep * 32 + t + hi * 8 + ((t >= 8) ? 8 : 0);
            bf[t] = (_Float16)((kb <= n) ? 1.0f : 0.0f);
        }
        c = __builtin_amdgcn_wmma_f32_16x16x32_f16(
                /*neg_a=*/false, a, /*neg_b=*/false, bf,
                /*c_mod=*/(short)0, c, /*reuse_a=*/false, /*reuse_b=*/false);
    }
#pragma unroll
    for (int r = 0; r < 8; ++r) {
        int row = mtile * 16 + r + 8 * hi;   // C/D: VGPR r -> M=r (lanes 0-15), M=r+8 (16-31)
        int col = ntile * 16 + l15;
        cdf[(size_t)row * NBINS + col] = c[r];
    }
}

// --------------------------------------------------------- CDF mse ----------
// cdf layout: 6 signals x 64 rows x 64 bins; pairs (0,1),(2,3),(4,5) -> acc[2..4]
__global__ void md_cdfdiff_kernel(const float* __restrict__ cdf, float* __restrict__ acc) {
    __shared__ float red[256];
    const int pair = blockIdx.x, tid = threadIdx.x;
    const float* cx = cdf + (size_t)(2 * pair) * (NB * NBINS);
    const float* cy = cx + (NB * NBINS);
    float s = 0.0f;
    for (int i = tid; i < NB * NBINS; i += 256) {
        float d = cx[i] - cy[i];
        s += d * d;
    }
    red[tid] = s; __syncthreads();
    for (int st = 128; st > 0; st >>= 1) { if (tid < st) red[tid] += red[tid + st]; __syncthreads(); }
    if (tid == 0) acc[2 + pair] = red[0] * (1.0f / (float)(NB * NBINS));
}

// --------------------------------------------------------- combine ----------
__global__ void md_final_kernel(const float* __restrict__ ws, float* __restrict__ out) {
    const float* acc = ws;        // [0]=|pred| [1]=|sd| [2]=cdf_t [3]=cdf_sd [4]=cdf_f [5]=freq
    const float* dt  = ws + 16;   // dtw_time[64]
    const float* ds  = ws + 80;   // dtw_sd[64]
    float mt = 0.0f, ms = 0.0f;
    for (int i = 0; i < NB; ++i) { mt += dt[i]; ms += ds[i]; }
    mt *= (1.0f / (float)NB);
    ms *= (1.0f / (float)NB);
    float l_time = mt + acc[0] + acc[2];
    float l_sd   = ms + acc[1] + acc[3];
    float l_freq = acc[5] + acc[4];
    float total  = 1.5f * l_time + 0.8f * l_freq + 1.2f * l_sd;
    out[0] = total; out[1] = l_time; out[2] = l_freq; out[3] = l_sd;
}

// ---------------------------------------------------------------------------
extern "C" void kernel_launch(void* const* d_in, const int* in_sizes, int n_in,
                              void* d_out, int out_size, void* d_ws, size_t ws_size,
                              hipStream_t stream) {
    (void)in_sizes; (void)n_in; (void)out_size; (void)ws_size;
    const float* pred = (const float*)d_in[0];   // (64,1,1024) f32
    const float* targ = (const float*)d_in[1];   // (64,1,1024) f32
    float* out = (float*)d_out;                  // 4 scalars

    // workspace layout (floats); total ~835,584 floats (~3.35 MB)
    float* ws     = (float*)d_ws;
    float* acc    = ws;                 // 16
    float* dtw_t  = ws + 16;            // 64
    float* dtw_s  = ws + 80;            // 64
    float* sdp    = ws + 256;           // 64*1022 = 65408
    float* sdt    = ws + 256 + 65408;   // 65408  (ends at 131072)
    float* fmp    = ws + 131072;        // 64*5*1024 = 327680
    float* fmt    = ws + 458752;        // 327680 (ends at 786432)
    float* hbuf   = ws + 786432;        // 6*64*64 = 24576
    float* cdfbuf = ws + 811008;        // 24576  (ends at 835584)

    md_init_acc<<<1, 16, 0, stream>>>(acc);

    md_sd_kernel<<<(NB * NTSD + 255) / 256, 256, 0, stream>>>(pred, targ, sdp, sdt);

    // soft-DTW wavefronts (one workgroup per batch, 32 waves each)
    md_dtw_kernel<<<NB, 1024, 0, stream>>>(pred, targ, NT,   NT,   dtw_t);
    md_dtw_kernel<<<NB, 1024, 0, stream>>>(sdp,  sdt,  NTSD, NTSD, dtw_s);

    md_absmean_kernel<<<64, 256, 0, stream>>>(pred, NB * NT,   &acc[0], 1.0f / (float)(NB * NT));
    md_absmean_kernel<<<64, 256, 0, stream>>>(sdp,  NB * NTSD, &acc[1], 1.0f / (float)(NB * NTSD));

    md_wavelet_kernel<<<NB, 256, 0, stream>>>(pred, fmp);
    md_wavelet_kernel<<<NB, 256, 0, stream>>>(targ, fmt);
    md_freq_kernel<<<NB, 256, 0, stream>>>(fmp, &acc[5]);

    // normalized soft histograms (6 signals x 64 rows)
    md_hist_kernel<<<NB, 256, 0, stream>>>(pred, NT,       NT,       hbuf + 0 * NB * NBINS);
    md_hist_kernel<<<NB, 256, 0, stream>>>(targ, NT,       NT,       hbuf + 1 * NB * NBINS);
    md_hist_kernel<<<NB, 256, 0, stream>>>(sdp,  NTSD,     NTSD,     hbuf + 2 * NB * NBINS);
    md_hist_kernel<<<NB, 256, 0, stream>>>(sdt,  NTSD,     NTSD,     hbuf + 3 * NB * NBINS);
    md_hist_kernel<<<NB, 256, 0, stream>>>(fmp,  NLVL * NT, NLVL * NT, hbuf + 4 * NB * NBINS);
    md_hist_kernel<<<NB, 256, 0, stream>>>(fmt,  NLVL * NT, NLVL * NT, hbuf + 5 * NB * NBINS);

    // cumsum -> CDF via WMMA: 384 rows -> 24 M-tiles x 4 N-tiles = 96 waves
    md_cumsum_wmma<<<96, 32, 0, stream>>>(hbuf, cdfbuf);

    md_cdfdiff_kernel<<<3, 256, 0, stream>>>(cdfbuf, acc);

    md_final_kernel<<<1, 1, 0, stream>>>(ws, out);
}
